// VecDGCNN_att_67508295958598
// MI455X (gfx1250) — matrix operand order repaired and checked by reading, not compiled, not run on hardware
//
#include <hip/hip_runtime.h>
#include <cstdint>

#define KNN 16
#define NEG_SLOPE 0.2f

typedef __attribute__((ext_vector_type(16))) __bf16 v16bf;
typedef __attribute__((ext_vector_type(8)))  float  v8f;
typedef __attribute__((ext_vector_type(2)))  float  v2f;

#if defined(__AMDGCN__) && __has_builtin(__builtin_amdgcn_wmma_f32_16x16x4_f32)
#define HAS_F32K4 1
#else
#define HAS_F32K4 0
#endif

__device__ __forceinline__ __bf16 f2bf(float f) { return (__bf16)f; }

__device__ __forceinline__ void wait_asynccnt0() {
#if defined(__gfx1250__)
#if __has_builtin(__builtin_amdgcn_s_wait_asynccnt)
  __builtin_amdgcn_s_wait_asynccnt(0);
#else
  asm volatile("s_wait_asynccnt 0x0" ::: "memory");
#endif
#endif
}

// ---------------------------------------------------------------------------
// Generic batched GEMM:  Y_b = W (MxK) * X_b (K x Ncols), fp32 in memory,
// bf16 WMMA f32-accumulate. Block = 128 threads (4 waves), output tile
// 16 (M) x 64 (N). The 32x64 fp32 B tile is staged through LDS with
// double-buffered global_load_async_to_lds_b128 (ASYNCcnt): issue tile t+1,
// compute tile t, one workgroup barrier per k-step.
// Fragment layouts per CDNA5 ISA 7.12.2 (wave32).
// ---------------------------------------------------------------------------
__global__ void gemm_wmma_kernel(const float* __restrict__ W,
                                 const float* __restrict__ X,
                                 float* __restrict__ Y,
                                 int M, int Kd, long long Ncols) {
  __shared__ float Bs[2][32][68];   // 68-float stride -> 16B-aligned rows
  const int b = blockIdx.z;
  const float* Xb = X + (long long)b * Kd * Ncols;
  float* Yb = Y + (long long)b * M * Ncols;
  const int tid = threadIdx.x;
  const int lane = tid & 31;
  const int wave = tid >> 5;
  const int m0 = blockIdx.y * 16;
  const long long nblk = (long long)blockIdx.x * 64;
  const long long n0 = nblk + (long long)wave * 16;

  // staging coords: thread copies rows {srow0, +8, +16, +24}, 4 floats each
  const int srow0 = tid >> 4;         // 0..7
  const int scol  = (tid & 15) * 4;   // 0..60

  v8f acc = {};
  const int  mrow   = m0 + (lane & 15);
  const int  kShift = (lane & 16) ? 8 : 0;
  const bool fullM  = (m0 + 16 <= M);
  const bool fullN  = (nblk + 64 <= Ncols) && ((Ncols & 3LL) == 0);
  const int  mclamp = (mrow < M) ? mrow : (M - 1);

  auto stage = [&](int k0, int buf) {
    if (fullN && (k0 + 32 <= Kd)) {
#if defined(__gfx1250__)
#pragma unroll
      for (int p = 0; p < 4; ++p) {
        const int r = srow0 + p * 8;
        const float* src = Xb + (long long)(k0 + r) * Ncols + nblk + scol;
        const unsigned ldsOff =
            (unsigned)(unsigned long long)(uintptr_t)&Bs[buf][r][scol];
        const unsigned long long ga = (unsigned long long)(uintptr_t)src;
        asm volatile("global_load_async_to_lds_b128 %0, %1, off"
                     :: "v"(ldsOff), "v"(ga) : "memory");
      }
#else
#pragma unroll
      for (int p = 0; p < 4; ++p) {
        const int r = srow0 + p * 8;
        const float* src = Xb + (long long)(k0 + r) * Ncols + nblk + scol;
#pragma unroll
        for (int e = 0; e < 4; ++e) Bs[buf][r][scol + e] = src[e];
      }
#endif
    } else {
      // branch-free edge staging: clamp addresses, select zero on OOB
#pragma unroll
      for (int p = 0; p < 4; ++p) {
        const int r = srow0 + p * 8;
        const int kk = k0 + r;
        const int kc = (kk < Kd) ? kk : (Kd - 1);
#pragma unroll
        for (int e = 0; e < 4; ++e) {
          const long long c = nblk + scol + e;
          const long long cc = (c < Ncols) ? c : (Ncols - 1);
          float v = Xb[(long long)kc * Ncols + cc];
          if (kk >= Kd || c >= Ncols) v = 0.0f;
          Bs[buf][r][scol + e] = v;
        }
      }
    }
  };

  stage(0, 0);
  int buf = 0;
  for (int k0 = 0; k0 < Kd; k0 += 32) {
    wait_asynccnt0();
    __syncthreads();
    if (k0 + 32 < Kd) stage(k0 + 32, buf ^ 1);

    // ---- A fragment: 16x32 bf16 of W (row-major MxK) ----
    v16bf a;
    if (fullM && (k0 + 32 <= Kd)) {
      const float* wr = W + (long long)mrow * Kd + k0 + kShift;
#pragma unroll
      for (int j = 0; j < 8; ++j) a[j] = f2bf(wr[j]);
#pragma unroll
      for (int j = 0; j < 8; ++j) a[8 + j] = f2bf(wr[16 + j]);
      if (k0 + 32 < Kd) __builtin_prefetch(wr + 32, 0, 1);
    } else {
      const float* wr = W + (long long)mclamp * Kd;
#pragma unroll
      for (int j = 0; j < 16; ++j) {
        const int kk = k0 + kShift + (j < 8 ? j : j + 8);
        const int kc = (kk < Kd) ? kk : (Kd - 1);
        float v = wr[kc];
        if (mrow >= M || kk >= Kd) v = 0.0f;
        a[j] = f2bf(v);
      }
    }
    // ---- B fragment from LDS: lane -> N column, 16 K values (same k order as A) ----
    v16bf bfr;
    const float* bsrow = &Bs[buf][kShift][wave * 16 + (lane & 15)];
#pragma unroll
    for (int e = 0; e < 16; ++e) bfr[e] = f2bf(bsrow[(e < 8 ? e : e + 8) * 68]);

    acc = __builtin_amdgcn_wmma_f32_16x16x32_bf16(false, a, false, bfr,
                                                  (short)0, acc, false, false);
    buf ^= 1;
  }

  // ---- store D ----
  const long long col = n0 + (lane & 15);
  const int rbase = m0 + ((lane & 16) ? 8 : 0);
  if (col < Ncols) {
    float* yp = Yb + (long long)rbase * Ncols + col;
    if (fullM) {
#pragma unroll
      for (int r = 0; r < 8; ++r) { *yp = acc[r]; yp += Ncols; }
    } else {
#pragma unroll
      for (int r = 0; r < 8; ++r) { if (rbase + r < M) *yp = acc[r]; yp += Ncols; }
    }
  }
}

// ---------------------------------------------------------------------------
// Pairwise squared distances via WMMA:
//   d2[b,m,n] = |dst_m|^2 + |src_n|^2 - 2 * dot(dst_m, src_n)
// dstF,srcF stored (B, D, N) row-major; Nd/Ns are multiples of 16 here,
// so only the K dimension needs guarding (branch-free clamps).
// ---------------------------------------------------------------------------
__global__ void pair_d2_kernel(const float* __restrict__ dstF,
                               const float* __restrict__ srcF,
                               const float* __restrict__ sqd,
                               const float* __restrict__ sqs,
                               float* __restrict__ d2,
                               int D, int Nd, int Ns) {
  const int b = blockIdx.z;
  const float* A  = dstF + (long long)b * D * Nd;  // A[m,k] = A[k*Nd+m]
  const float* Bm = srcF + (long long)b * D * Ns;  // B[k,n] = Bm[k*Ns+n]
  const int lane = threadIdx.x & 31;
  const int wave = threadIdx.x >> 5;
  const int m0 = blockIdx.y * 16;
  const int n0 = blockIdx.x * 64 + wave * 16;
  if (n0 >= Ns) return;
  v8f acc = {};
#if HAS_F32K4
  {
    const int m  = m0 + (lane & 15);
    const int ko = (lane & 16) ? 2 : 0;
    const int n  = n0 + (lane & 15);
    for (int k0 = 0; k0 < D; k0 += 4) {
      const int ka = k0 + ko, kb = k0 + ko + 1;
      const int kac = (ka < D) ? ka : (D - 1);
      const int kbc = (kb < D) ? kb : (D - 1);
      v2f a, bb;
      float a0 = A[(long long)kac * Nd + m];
      float a1 = A[(long long)kbc * Nd + m];
      float b0 = Bm[(long long)kac * Ns + n];
      float b1 = Bm[(long long)kbc * Ns + n];
      if (ka >= D) { a0 = 0.f; b0 = 0.f; }
      if (kb >= D) { a1 = 0.f; b1 = 0.f; }
      a[0] = a0; a[1] = a1; bb[0] = b0; bb[1] = b1;
      acc = __builtin_amdgcn_wmma_f32_16x16x4_f32(false, a, false, bb,
                                                  (short)0, acc, false, false);
    }
  }
#else
  {
    const int mrow = m0 + (lane & 15);
    const int kShift = (lane & 16) ? 8 : 0;
    for (int k0 = 0; k0 < D; k0 += 32) {
      v16bf a, bfr;
#pragma unroll
      for (int j = 0; j < 16; ++j) {
        const int kk = k0 + kShift + (j < 8 ? j : j + 8);
        const int kc = (kk < D) ? kk : (D - 1);
        float v = A[(long long)kc * Nd + mrow];
        if (kk >= D) v = 0.f;
        a[j] = f2bf(v);
      }
#pragma unroll
      for (int e = 0; e < 16; ++e) {
        const int kb = k0 + kShift + (e < 8 ? e : e + 8);
        const int kbc = (kb < D) ? kb : (D - 1);
        float v = Bm[(long long)kbc * Ns + n0 + (lane & 15)];
        if (kb >= D) v = 0.f;
        bfr[e] = f2bf(v);
      }
      acc = __builtin_amdgcn_wmma_f32_16x16x32_bf16(false, a, false, bfr,
                                                    (short)0, acc, false, false);
    }
  }
#endif
  const int col = n0 + (lane & 15);
  const int rbase = m0 + ((lane & 16) ? 8 : 0);
  float* dp = d2 + ((long long)b * Nd + rbase) * Ns + col;
  const float sc = sqs[(long long)b * Ns + col];
  const float* sdp = sqd + (long long)b * Nd + rbase;
#pragma unroll
  for (int r = 0; r < 8; ++r) {
    *dp = sdp[r] + sc - 2.0f * acc[r];
    dp += Ns;
  }
}

// per-column squared norms of (B, D, N) tensor
__global__ void sqnorm_kernel(const float* __restrict__ f, float* __restrict__ out,
                              int D, int N, long long total) {
  long long t = (long long)blockIdx.x * blockDim.x + threadIdx.x;
  if (t >= total) return;
  const int n = (int)(t % N);
  const int b = (int)(t / N);
  const float* p = f + (long long)b * D * N + n;
  float s = 0.f;
  for (int d = 0; d < D; ++d) { float v = p[(long long)d * N]; s += v * v; }
  out[t] = s;
}

// iterative top-16 min (ascending, lowest-index tie-break), 1 wave per row,
// candidates held in registers. Ns is always a multiple of 32 here.
__global__ void topk16_kernel(const float* __restrict__ d2, int* __restrict__ idx, int Ns) {
  const int row = blockIdx.x;
  const float* r = d2 + (long long)row * Ns;
  const int lane = threadIdx.x;
  float loc[64];
  const int cnt = Ns >> 5;
  for (int t = 0; t < cnt; ++t) loc[t] = r[lane + t * 32];
  int* out = idx + (long long)row * KNN;
  for (int it = 0; it < KNN; ++it) {
    float bv = 3.0e38f; int bi = 0x7fffffff;
    for (int t = 0; t < cnt; ++t) {
      const int j = lane + t * 32;
      if (loc[t] < bv) { bv = loc[t]; bi = j; }
    }
    for (int off = 16; off > 0; off >>= 1) {
      const float ov = __shfl_xor(bv, off);
      const int   oi = __shfl_xor(bi, off);
      if (ov < bv || (ov == bv && oi < bi)) { bv = ov; bi = oi; }
    }
    if (lane == (bi & 31)) loc[bi >> 5] = 3.0e38f;
    if (lane == 0) out[it] = bi;
  }
}

// farthest point sampling, one block per batch; xyz (B,3,N)
__global__ void fps_kernel(const float* __restrict__ xyz, int* __restrict__ out,
                           int N, int Kout) {
  __shared__ float dist[2048];
  __shared__ float rv[256];
  __shared__ int   ri[256];
  __shared__ int   curIdx;
  const int b = blockIdx.x, t = threadIdx.x;
  const float* X = xyz + (long long)b * 3 * N;
  for (int n = t; n < N; n += 256) dist[n] = 1e10f;
  if (t == 0) curIdx = 0;
  __syncthreads();
  for (int k = 0; k < Kout; ++k) {
    const int last = curIdx;
    if (t == 0) out[(long long)b * Kout + k] = last;
    const float px = X[last], py = X[N + last], pz = X[2 * N + last];
    float bv = -1.f; int bi = 0;
    for (int n = t; n < N; n += 256) {
      const float dx = X[n] - px, dy = X[N + n] - py, dz = X[2 * N + n] - pz;
      const float nd = fminf(dist[n], dx * dx + dy * dy + dz * dz);
      dist[n] = nd;
      if (nd > bv) { bv = nd; bi = n; }
    }
    rv[t] = bv; ri[t] = bi;
    __syncthreads();
    for (int s = 128; s > 0; s >>= 1) {
      if (t < s) {
        if (rv[t + s] > rv[t] || (rv[t + s] == rv[t] && ri[t + s] < ri[t])) {
          rv[t] = rv[t + s]; ri[t] = ri[t + s];
        }
      }
      __syncthreads();
    }
    if (t == 0) curIdx = ri[0];
    __syncthreads();
  }
}

// out[(b*rows+r)*Nd+n] = in[(b*rows+r)*Ns + idx[b*Nd+n]]
__global__ void gather_rows_kernel(const float* __restrict__ in, float* __restrict__ out,
                                   const int* __restrict__ idx, int rows, int Ns, int Nd,
                                   long long total) {
  long long t = (long long)blockIdx.x * blockDim.x + threadIdx.x;
  if (t >= total) return;
  const int n = (int)(t % Nd);
  long long br = t / Nd;
  const int b = (int)(br / rows);
  out[t] = in[br * Ns + idx[(long long)b * Nd + n]];
}

// layer-0 graph feature (C=1, cross product variant): y (B,3,3,N,KNN)
__global__ void graph_feat0_kernel(const float* __restrict__ srcF,
                                   const int* __restrict__ idx,
                                   float* __restrict__ y, int N, long long total) {
  long long t = (long long)blockIdx.x * blockDim.x + threadIdx.x;
  if (t >= total) return;
  const int k = (int)(t % KNN);
  long long u = t / KNN;
  const int n = (int)(u % N);
  const int b = (int)(u / N);
  const int j = idx[((long long)b * N + n) * KNN + k];
  const float* S = srcF + (long long)b * 3 * N;
  const float s0 = S[j], s1 = S[N + j], s2 = S[2 * N + j];
  const float d0 = S[n], d1 = S[N + n], d2v = S[2 * N + n];
  float nr = sqrtf(d0 * d0 + d1 * d1 + d2v * d2v);
  nr = fmaxf(nr, 1e-12f);
  const float x0 = d0 / nr, x1 = d1 / nr, x2 = d2v / nr;
  const float c0 = x1 * s2 - x2 * s1;
  const float c1 = x2 * s0 - x0 * s2;
  const float c2 = x0 * s1 - x1 * s0;
  const long long yb = (long long)b * 3 * 3 * N * KNN;
  const long long nk = (long long)n * KNN + k;
  const long long NK = (long long)N * KNN;
  y[yb + 0 * NK + nk] = c0;  y[yb + 1 * NK + nk] = c1;  y[yb + 2 * NK + nk] = c2;
  y[yb + 3 * NK + nk] = s0 - d0; y[yb + 4 * NK + nk] = s1 - d1; y[yb + 5 * NK + nk] = s2 - d2v;
  y[yb + 6 * NK + nk] = d0;  y[yb + 7 * NK + nk] = d1;  y[yb + 8 * NK + nk] = d2v;
}

// general graph feature: y[:, :C] = nn - dst ; y[:, C:] = dst
__global__ void graph_feat_kernel(const float* __restrict__ srcF,
                                  const float* __restrict__ dstF,
                                  const int* __restrict__ idx,
                                  float* __restrict__ y,
                                  int C, int Ns, int Nd, long long total) {
  long long t = (long long)blockIdx.x * blockDim.x + threadIdx.x;
  if (t >= total) return;   // total = B*C*3*Nd*KNN
  const int k = (int)(t % KNN);
  long long u = t / KNN;
  const int n = (int)(u % Nd); u /= Nd;
  const int x = (int)(u % 3);  u /= 3;
  const int c = (int)(u % C);
  const int b = (int)(u / C);
  const int j = idx[((long long)b * Nd + n) * KNN + k];
  const float nn = srcF[(((long long)b * C + c) * 3 + x) * Ns + j];
  const float dv = dstF[(((long long)b * C + c) * 3 + x) * Nd + n];
  const long long yb = (long long)b * (2 * C) * 3 * Nd * KNN;
  const long long o1 = yb + ((((long long)c) * 3 + x) * Nd + n) * KNN + k;
  const long long o2 = yb + ((((long long)(C + c)) * 3 + x) * Nd + n) * KNN + k;
  y[o1] = nn - dv;
  y[o2] = dv;
}

// vector-neuron activation; tensors (B,C,3,Ninner)
__global__ void vn_act_kernel(const float* __restrict__ q, const float* __restrict__ d,
                              float* __restrict__ out, long long Ninner, long long total) {
  long long t = (long long)blockIdx.x * blockDim.x + threadIdx.x;
  if (t >= total) return;   // total = B*C*Ninner
  const long long n = t % Ninner;
  const long long bc = t / Ninner;
  const long long base = bc * 3 * Ninner + n;
  const float q0 = q[base], q1 = q[base + Ninner], q2 = q[base + 2 * Ninner];
  const float d0 = d[base], d1 = d[base + Ninner], d2v = d[base + 2 * Ninner];
  const float dot = q0 * d0 + q1 * d1 + q2 * d2v;
  const float dsq = d0 * d0 + d1 * d1 + d2v * d2v + 1e-6f;
  const float s = (NEG_SLOPE - 1.0f) * fminf(dot, 0.0f) / dsq;
  out[base]              = q0 + s * d0;
  out[base + Ninner]     = q1 + s * d1;
  out[base + 2 * Ninner] = q2 + s * d2v;
}

__global__ void mean_k_kernel(const float* __restrict__ in, float* __restrict__ out,
                              long long total) {
  long long t = (long long)blockIdx.x * blockDim.x + threadIdx.x;
  if (t >= total) return;
  const float* p = in + t * KNN;
  float s = 0.f;
#pragma unroll
  for (int k = 0; k < KNN; ++k) s += p[k];
  out[t] = s * (1.0f / KNN);
}

// cevn: x (B,C,3,Ninner) -> x/max(|x|,eps) * |x|/max(||.||_C, eps)
__global__ void cevn_norm_kernel(const float* __restrict__ x, float* __restrict__ n,
                                 long long Ninner, long long total) {
  long long t = (long long)blockIdx.x * blockDim.x + threadIdx.x;
  if (t >= total) return;
  const long long inn = t % Ninner;
  const long long bc = t / Ninner;
  const long long base = bc * 3 * Ninner + inn;
  const float a = x[base], b = x[base + Ninner], c = x[base + 2 * Ninner];
  n[t] = sqrtf(a * a + b * b + c * c);
}
__global__ void cevn_cnorm_kernel(const float* __restrict__ n, float* __restrict__ r,
                                  int C, long long Ninner, long long total) {
  long long t = (long long)blockIdx.x * blockDim.x + threadIdx.x;
  if (t >= total) return;  // total = B*Ninner
  const long long inn = t % Ninner;
  const long long b = t / Ninner;
  float s = 0.f;
  for (int c = 0; c < C; ++c) {
    const float v = n[((long long)b * C + c) * Ninner + inn];
    s += v * v;
  }
  r[t] = sqrtf(s);
}
__global__ void cevn_apply_kernel(const float* __restrict__ x, const float* __restrict__ n,
                                  const float* __restrict__ r, float* __restrict__ out,
                                  int C, long long Ninner, long long total) {
  long long t = (long long)blockIdx.x * blockDim.x + threadIdx.x;
  if (t >= total) return;  // total = B*C*Ninner
  const long long inn = t % Ninner;
  const long long bc = t / Ninner;
  const long long b = bc / C;
  const float nn = n[t];
  const float s = nn / (fmaxf(nn, 1e-12f) * fmaxf(r[b * Ninner + inn], 1e-12f));
  const long long base = bc * 3 * Ninner + inn;
  out[base]              = x[base] * s;
  out[base + Ninner]     = x[base + Ninner] * s;
  out[base + 2 * Ninner] = x[base + 2 * Ninner] * s;
}

__global__ void qk_kernel(const float* __restrict__ kf, const float* __restrict__ qf,
                          float* __restrict__ qk, int Nd, long long total) {
  long long t = (long long)blockIdx.x * blockDim.x + threadIdx.x;
  if (t >= total) return;  // total = B*C*Nd*KNN
  const int k = (int)(t % KNN);
  long long u = t / KNN;
  const int n = (int)(u % Nd);
  const long long bc = u / Nd;
  const long long NK = (long long)Nd * KNN;
  const long long kb = bc * 3 * NK + (long long)n * KNN + k;
  const long long qb = bc * 3 * Nd + n;
  qk[t] = kf[kb] * qf[qb] + kf[kb + NK] * qf[qb + Nd] + kf[kb + 2 * NK] * qf[qb + 2 * Nd];
}

__global__ void att_softmax_kernel(const float* __restrict__ qk, float* __restrict__ att,
                                   int C, int Nd, long long total) {
  long long t = (long long)blockIdx.x * blockDim.x + threadIdx.x;
  if (t >= total) return;  // total = B*H*Nd
  const int H = C / 16;
  const int n = (int)(t % Nd);
  long long u = t / Nd;
  const int h = (int)(u % H);
  const int b = (int)(u / H);
  float s[KNN];
#pragma unroll
  for (int k = 0; k < KNN; ++k) s[k] = 0.f;
  for (int m = 0; m < 16; ++m) {
    const long long base = (((long long)b * C + h * 16 + m) * Nd + n) * KNN;
#pragma unroll
    for (int k = 0; k < KNN; ++k) s[k] += qk[base + k];
  }
  float mx = -3.0e38f;
#pragma unroll
  for (int k = 0; k < KNN; ++k) { s[k] *= 0.14433756729740643f; mx = fmaxf(mx, s[k]); }
  float se = 0.f;
#pragma unroll
  for (int k = 0; k < KNN; ++k) { s[k] = expf(s[k] - mx); se += s[k]; }
  const long long ob = (((long long)b * H + h) * Nd + n) * KNN;
#pragma unroll
  for (int k = 0; k < KNN; ++k) att[ob + k] = s[k] / se;
}

__global__ void att_apply_kernel(const float* __restrict__ att, const float* __restrict__ v,
                                 float* __restrict__ out, int C, int Nd, long long total) {
  long long t = (long long)blockIdx.x * blockDim.x + threadIdx.x;
  if (t >= total) return;  // total = B*C*3*Nd ; out (B,C,3,Nd)
  const int n = (int)(t % Nd);
  long long u = t / Nd;
  const int x = (int)(u % 3); u /= 3;
  const int c = (int)(u % C);
  const int b = (int)(u / C);
  const int H = C / 16;
  const float* ap = att + (((long long)b * H + c / 16) * Nd + n) * KNN;
  const float* vp = v + ((((long long)b * C + c) * 3 + x) * Nd + n) * KNN;
  float s = 0.f;
#pragma unroll
  for (int k = 0; k < KNN; ++k) s += ap[k] * vp[k];
  out[t] = s;
}

__global__ void mean_n_kernel(const float* __restrict__ in, float* __restrict__ out,
                              int Nd, long long total) {
  long long t = (long long)blockIdx.x * blockDim.x + threadIdx.x;
  if (t >= total) return;  // total = B*C*3
  float s = 0.f;
  const float* p = in + t * (long long)Nd;
  for (int n = 0; n < Nd; ++n) s += p[n];
  out[t] = s / (float)Nd;
}

__global__ void cat_kernel(const float* __restrict__ f, const float* __restrict__ g,
                           float* __restrict__ cat, int C, int Nd, long long total) {
  long long t = (long long)blockIdx.x * blockDim.x + threadIdx.x;
  if (t >= total) return;  // total = B*2C*3*Nd
  const int n = (int)(t % Nd);
  long long u = t / Nd;
  const int x = (int)(u % 3); u /= 3;
  const int c2 = (int)(u % (2 * C));
  const int b = (int)(u / (2 * C));
  cat[t] = (c2 < C) ? f[(((long long)b * C + c2) * 3 + x) * Nd + n]
                    : g[((long long)b * C + (c2 - C)) * 3 + x];
}

__global__ void scale_kernel(const float* __restrict__ yv, float* __restrict__ out, int C) {
  __shared__ float sh[256];
  const int b = blockIdx.x, t = threadIdx.x;
  float v = 0.f;
  if (t < C) {
    const float* p = yv + ((long long)b * C + t) * 3;
    v = sqrtf(p[0] * p[0] + p[1] * p[1] + p[2] * p[2]);
  }
  sh[t] = v;
  __syncthreads();
  for (int s = 128; s > 0; s >>= 1) {
    if (t < s) sh[t] += sh[t + s];
    __syncthreads();
  }
  if (t == 0) out[b] = sh[0] / (float)C * 640.0f;
}

__global__ void add_kernel(const float* a, const float* b, float* o, long long n) {
  long long t = (long long)blockIdx.x * blockDim.x + threadIdx.x;
  if (t < n) o[t] = a[t] + b[t];
}
__global__ void dot3_kernel(const float* a, const float* s, float* o, long long n) {
  long long t = (long long)blockIdx.x * blockDim.x + threadIdx.x;
  if (t >= n) return;
  const float* pa = a + t * 3; const float* ps = s + t * 3;
  o[t] = pa[0] * ps[0] + pa[1] * ps[1] + pa[2] * ps[2];
}
__global__ void copy_kernel(const float* a, float* o, long long n) {
  long long t = (long long)blockIdx.x * blockDim.x + threadIdx.x;
  if (t < n) o[t] = a[t];
}

// ---------------------------------------------------------------------------

#define GRID(n) dim3((unsigned)(((n) + 255) / 256))

extern "C" void kernel_launch(void* const* d_in, const int* in_sizes, int n_in,
                              void* d_out, int out_size, void* d_ws, size_t ws_size,
                              hipStream_t stream) {
  (void)in_sizes; (void)n_in; (void)out_size; (void)ws_size;
  const int B = 4, N0 = 2048;
  const int FEAT[8] = {32, 32, 64, 64, 128, 256, 512, 512};
  const bool DOWN[8] = {false, false, true, false, true, false, true, false};

  const float* x = (const float*)d_in[0];
  int pi = 1;
  const float *VW[8], *VU[8], *QW[8], *QU[8], *KW[8], *KU[8], *GW[8], *GU[8];
  for (int i = 0; i < 8; ++i) {
    VW[i] = (const float*)d_in[pi++]; VU[i] = (const float*)d_in[pi++];
    if (i >= 2) {
      QW[i] = (const float*)d_in[pi++]; QU[i] = (const float*)d_in[pi++];
      KW[i] = (const float*)d_in[pi++]; KU[i] = (const float*)d_in[pi++];
      GW[i] = (const float*)d_in[pi++]; GU[i] = (const float*)d_in[pi++];
    }
  }
  const float* CcW = (const float*)d_in[pi++]; const float* CcU = (const float*)d_in[pi++];
  const float* M0W = (const float*)d_in[pi++]; const float* M0U = (const float*)d_in[pi++];
  const float* M1W = (const float*)d_in[pi++]; const float* M1U = (const float*)d_in[pi++];
  const float* L0W = (const float*)d_in[pi++]; const float* L0U = (const float*)d_in[pi++];
  const float* L1W = (const float*)d_in[pi++]; const float* L1U = (const float*)d_in[pi++];
  const float* ImW = (const float*)d_in[pi++]; const float* IlW = (const float*)d_in[pi++];
  const float* CtW0 = (const float*)d_in[pi++]; const float* CtU0 = (const float*)d_in[pi++];
  const float* CtW1 = (const float*)d_in[pi++]; const float* CtWs = (const float*)d_in[pi++];

  // ---- workspace arena (deterministic layout) ----
  char* wsBase = (char*)d_ws;
  size_t wsOff = 0;
  auto allocF = [&](long long n) -> float* {
    float* p = (float*)(wsBase + wsOff);
    wsOff += (((size_t)n * 4) + 255) & ~(size_t)255;
    return p;
  };
  auto allocI = [&](long long n) -> int* {
    int* p = (int*)(wsBase + wsOff);
    wsOff += (((size_t)n * 4) + 255) & ~(size_t)255;
    return p;
  };
  float* xyzA = allocF(B * 3 * N0);
  float* xyzB = allocF(B * 3 * N0);
  float* fA = allocF(1 << 20); float* fB = allocF(1 << 20); float* fC = allocF(1 << 20);
  float* sqS = allocF(B * N0); float* sqD = allocF(B * N0);
  float* qfq = allocF(524288); float* qfd = allocF(524288); float* cnq = allocF(262144);
  float* rbuf = allocF(131072);
  float* qkb = allocF(2228224);
  float* attb = allocF(524288);
  float* gbuf = allocF(8192);
  float* catb = allocF(1 << 20);
  float* gq = allocF(524288); float* gd = allocF(524288);
  float* kfb = allocF(6815744);
  float* hq = allocF(98304); float* hd = allocF(98304);
  float* yv = allocF(3072); float* so3b = allocF(3072);
  float* ta = allocF(3072); float* tb = allocF(3072); float* tc = allocF(3072);
  float* meanb = allocF(3072); float* lvb = allocF(3072); float* smb = allocF(3072);
  float* t128a = allocF(1536); float* t128b = allocF(1536);
  float* c0b = allocF(64); float* c1b = allocF(64);
  int* idxK = allocI(B * N0 * KNN);
  int* idxF = allocI(B * 512);
  float* big0 = allocF(26214400);  // d2 / y (reused sequentially)
  float* big1 = allocF(13107200);  // q / v
  float* big2 = allocF(13107200);  // d / kf

  auto gemm = [&](const float* Wm, const float* Xm, float* Ym, int M, int Kd, long long Ncols) {
    dim3 g((unsigned)((Ncols + 63) / 64), (unsigned)((M + 15) / 16), (unsigned)B);
    gemm_wmma_kernel<<<g, 128, 0, stream>>>(Wm, Xm, Ym, M, Kd, Ncols);
  };
  auto lna = [&](const float* Wm, const float* Um, const float* Xin, float* qb, float* db,
                 float* ob, int M, int Kd, long long Ninner) {
    gemm(Wm, Xin, qb, M, Kd, 3 * Ninner);
    gemm(Um, qb, db, M, M, 3 * Ninner);
    const long long tot = (long long)B * M * Ninner;
    vn_act_kernel<<<GRID(tot), 256, 0, stream>>>(qb, db, ob, Ninner, tot);
  };
  auto cevnF = [&](float* Xm, float* nb, float* rb, int C, long long Ninner) {
    const long long tot = (long long)B * C * Ninner;
    cevn_norm_kernel<<<GRID(tot), 256, 0, stream>>>(Xm, nb, Ninner, tot);
    const long long tbn = (long long)B * Ninner;
    cevn_cnorm_kernel<<<GRID(tbn), 256, 0, stream>>>(nb, rb, C, Ninner, tbn);
    cevn_apply_kernel<<<GRID(tot), 256, 0, stream>>>(Xm, nb, rb, Xm, C, Ninner, tot);
  };

  // ---- init: src_xyz = src_f = x (B,3,2048) ----
  hipMemcpyAsync(xyzA, x, sizeof(float) * B * 3 * N0, hipMemcpyDeviceToDevice, stream);
  hipMemcpyAsync(fA, x, sizeof(float) * B * 3 * N0, hipMemcpyDeviceToDevice, stream);

  float* curX = xyzA; float* freeX = xyzB;
  float* curF = fA; float* free1 = fB; float* free2 = fC;
  int Cs = 1, Ns = N0;

  for (int i = 0; i < 8; ++i) {
    const int Co = FEAT[i];
    const int Nd = DOWN[i] ? Ns / 4 : Ns;
    float *dstX, *dstF, *newF;
    if (DOWN[i]) {
      fps_kernel<<<dim3(B), 256, 0, stream>>>(curX, idxF, Ns, Nd);
      dstX = freeX; dstF = free1; newF = free2;
      gather_rows_kernel<<<GRID((long long)B * 3 * Nd), 256, 0, stream>>>(
          curX, dstX, idxF, 3, Ns, Nd, (long long)B * 3 * Nd);
      gather_rows_kernel<<<GRID((long long)B * Cs * 3 * Nd), 256, 0, stream>>>(
          curF, dstF, idxF, Cs * 3, Ns, Nd, (long long)B * Cs * 3 * Nd);
    } else {
      dstX = curX; dstF = curF; newF = free1;
    }
    const int D = Cs * 3;
    sqnorm_kernel<<<GRID((long long)B * Ns), 256, 0, stream>>>(curF, sqS, D, Ns, (long long)B * Ns);
    sqnorm_kernel<<<GRID((long long)B * Nd), 256, 0, stream>>>(dstF, sqD, D, Nd, (long long)B * Nd);
    {
      dim3 g((unsigned)((Ns + 63) / 64), (unsigned)((Nd + 15) / 16), (unsigned)B);
      pair_d2_kernel<<<g, 128, 0, stream>>>(dstF, curF, sqD, sqS, big0, D, Nd, Ns);
    }
    topk16_kernel<<<dim3((unsigned)(B * Nd)), 32, 0, stream>>>(big0, idxK, Ns);

    int Cy; float* y = big0;  // y overwrites d2 (done with it)
    if (i == 0) {
      Cy = 3;
      const long long tot = (long long)B * Ns * KNN;
      graph_feat0_kernel<<<GRID(tot), 256, 0, stream>>>(curF, idxK, y, Ns, tot);
    } else {
      Cy = 2 * Cs;
      const long long tot = (long long)B * Cs * 3 * Nd * KNN;
      graph_feat_kernel<<<GRID(tot), 256, 0, stream>>>(curF, dstF, idxK, y, Cs, Ns, Nd, tot);
    }
    const long long NiYK = (long long)Nd * KNN;

    if (i < 2) {
      lna(VW[i], VU[i], y, big1, big2, big1, Co, Cy, NiYK);
      const long long tot = (long long)B * Co * 3 * Nd;
      mean_k_kernel<<<GRID(tot), 256, 0, stream>>>(big1, newF, tot);
    } else {
      lna(VW[i], VU[i], y, big1, big2, big1, Co, Cy, NiYK);   // v -> big1
      lna(KW[i], KU[i], y, big2, kfb, big2, Co, Cy, NiYK);    // kf -> big2
      cevnF(big2, kfb, rbuf, Co, NiYK);
      lna(QW[i], QU[i], dstF, qfq, qfd, qfq, Co, Cs, (long long)Nd);
      cevnF(qfq, cnq, rbuf, Co, (long long)Nd);
      const long long totQK = (long long)B * Co * Nd * KNN;
      qk_kernel<<<GRID(totQK), 256, 0, stream>>>(big2, qfq, qkb, Nd, totQK);
      const long long totSM = (long long)B * (Co / 16) * Nd;
      att_softmax_kernel<<<GRID(totSM), 256, 0, stream>>>(qkb, attb, Co, Nd, totSM);
      const long long totAP = (long long)B * Co * 3 * Nd;
      att_apply_kernel<<<GRID(totAP), 256, 0, stream>>>(attb, big1, newF, Co, Nd, totAP);
    }
    if (i >= 2) {
      const long long totG = (long long)B * Co * 3;
      mean_n_kernel<<<GRID(totG), 256, 0, stream>>>(newF, gbuf, Nd, totG);
      const long long totC = (long long)B * 2 * Co * 3 * Nd;
      cat_kernel<<<GRID(totC), 256, 0, stream>>>(newF, gbuf, catb, Co, Nd, totC);
      lna(GW[i], GU[i], catb, gq, gd, newF, Co, 2 * Co, (long long)Nd);
    }
    if (DOWN[i]) {
      float* t = curX; curX = dstX; freeX = t;
      float* o = curF; curF = newF; free2 = o;
    } else {
      float* o = curF; curF = newF; free1 = o;
    }
    Cs = Co; Ns = Nd;
  }

  // ---- head (curF: (B,512,3,32)) ----
  float* outF = (float*)d_out;
  lna(CcW, CcU, curF, hq, hd, hq, 256, 512, (long long)Ns);   // xc (B,256,3,32)
  mean_n_kernel<<<GRID((long long)B * 256 * 3), 256, 0, stream>>>(hq, yv, Ns, (long long)B * 256 * 3);
  copy_kernel<<<GRID(3072), 256, 0, stream>>>(yv, so3b, 3072);
  cevnF(so3b, ta, rbuf, 256, 1);                              // so3 == yn
  copy_kernel<<<GRID(3072), 256, 0, stream>>>(so3b, outF + 2048, 3072);
  scale_kernel<<<dim3(B), 256, 0, stream>>>(yv, outF + 5120, 256);
  // center
  lna(CtW0, CtU0, yv, t128a, t128b, t128a, 128, 256, 1);
  gemm(CtWs, yv, c0b, 1, 256, 3);
  gemm(CtW1, t128a, c1b, 1, 128, 3);
  add_kernel<<<GRID(12), 256, 0, stream>>>(c0b, c1b, outF + 5124, 12);
  // mean branch
  lna(M0W, M0U, so3b, ta, tb, ta, 256, 256, 1);
  lna(M1W, M1U, ta, tb, tc, meanb, 256, 256, 1);
  gemm(ImW, meanb, smb, 256, 256, 3);
  dot3_kernel<<<GRID((long long)B * 256), 256, 0, stream>>>(meanb, smb, outF, (long long)B * 256);
  // logvar branch
  lna(L0W, L0U, so3b, ta, tb, ta, 256, 256, 1);
  lna(L1W, L1U, ta, tb, tc, lvb, 256, 256, 1);
  gemm(IlW, lvb, smb, 256, 256, 3);
  dot3_kernel<<<GRID((long long)B * 256), 256, 0, stream>>>(lvb, smb, outF + 1024, (long long)B * 256);
}
